// LigRecGVP_11948599017846
// MI455X (gfx1250) — matrix operand order repaired and checked by reading, not compiled, not run on hardware
//
#include <hip/hip_runtime.h>
#include <hip/hip_bf16.h>
#include <cstdint>

// ---------------------------------------------------------------------------
// Types
// ---------------------------------------------------------------------------
typedef __bf16 bf16_t;
typedef __attribute__((ext_vector_type(16))) __bf16 v16bf;
typedef __attribute__((ext_vector_type(8)))  float  v8f;

#define NLIG 20000
#define NKP  4000
#define ELL  320000
#define EKL  80000
#define SDIM 256
#define VDIM 16

static inline int ceil_div(int a, int b) { return (a + b - 1) / b; }

__device__ __forceinline__ bf16_t f2bf(float x) {
  union { float f; unsigned u; } c; c.f = x;
  unsigned r = c.u + 0x7FFFu + ((c.u >> 16) & 1u);   // round-to-nearest-even
  unsigned short h = (unsigned short)(r >> 16);
  union { unsigned short s; bf16_t b; } o; o.s = h;
  return o.b;
}

__device__ __forceinline__ float silu_f(float x) { return x / (1.f + __expf(-x)); }

__device__ __forceinline__ uint32_t lds_off(const void* p) {
  // generic LDS addresses carry the LDS byte offset in the low 32 bits
  return (uint32_t)(uintptr_t)p;
}

__device__ __forceinline__ void async_b128(uint32_t dst_lds, const void* gaddr) {
  uint64_t ga = (uint64_t)(uintptr_t)gaddr;
  asm volatile("global_load_async_to_lds_b128 %0, %1, off"
               :: "v"(dst_lds), "v"(ga) : "memory");
}

__device__ __forceinline__ void wait_async0() {
  asm volatile("s_wait_asynccnt 0x0" ::: "memory");
}

// ---------------------------------------------------------------------------
// WMMA GEMM:  C[M,N] = A[M,K](bf16) @ B[N,K](bf16)^T  (+bias), optional SiLU copy
// K must be a multiple of 32.  Block tile 128x64, 8 waves, wave tile 16x64
// (4 accumulators). Double-buffered LDS fed by async global->LDS DMA loads.
// ---------------------------------------------------------------------------
#define BM 128
#define BN 64
#define BK 32

__global__ __launch_bounds__(256, 2)
void gemm_xwt_kernel(const bf16_t* __restrict__ A, const bf16_t* __restrict__ B,
                     const float* __restrict__ bias,
                     float* __restrict__ C, float* __restrict__ Csilu,
                     int M, int N, int K, int ldc)
{
  __shared__ bf16_t As[2][BM][BK];   // 2 x 8 KB
  __shared__ bf16_t Bs[2][BN][BK];   // 2 x 4 KB

  const int tid  = threadIdx.x;
  const int wave = tid >> 5;
  const int lane = tid & 31;
  const int m0 = blockIdx.x * BM;
  const int n0 = blockIdx.y * BN;
  const int wm = wave * 16;

  v8f acc0 = {}, acc1 = {}, acc2 = {}, acc3 = {};

  // cooperative tile-load mapping: 256 threads, 16B each per row-chunk
  const int lr = tid >> 2;          // 0..63
  const int lc = (tid & 3) * 8;     // 0,8,16,24 (bf16 elements)

  // clamp out-of-range rows (results for those rows/cols are never stored;
  // K is exactly zero-padded in the packed activation buffers)
  int ra0 = m0 + lr;      if (ra0 >= M) ra0 = M - 1;
  int ra1 = m0 + 64 + lr; if (ra1 >= M) ra1 = M - 1;
  int rb  = n0 + lr;      if (rb  >= N) rb  = N - 1;

  const bf16_t* pa0 = A + (size_t)ra0 * K + lc;
  const bf16_t* pa1 = A + (size_t)ra1 * K + lc;
  const bf16_t* pb  = B + (size_t)rb  * K + lc;

  uint32_t lA0[2], lA1[2], lB[2];
  lA0[0] = lds_off(&As[0][lr][lc]);      lA0[1] = lds_off(&As[1][lr][lc]);
  lA1[0] = lds_off(&As[0][64 + lr][lc]); lA1[1] = lds_off(&As[1][64 + lr][lc]);
  lB[0]  = lds_off(&Bs[0][lr][lc]);      lB[1]  = lds_off(&Bs[1][lr][lc]);

  const int nk = K >> 5;

  // prologue: async DMA tile 0 into buffer 0
  async_b128(lA0[0], pa0);
  async_b128(lA1[0], pa1);
  async_b128(lB[0],  pb);
  wait_async0();
  __syncthreads();

  const int am  = wm + (lane & 15);
  const int alo = (lane >> 4) * 16;     // contiguous k = 16h + s (matches B)
  const int bn   = lane & 15;
  const int bklo = (lane >> 4) * 16;    // contiguous k = 16h + s (matches A)

  for (int t = 0; t < nk; ++t) {
    const int buf = t & 1;
    if (t + 1 < nk) {
      const int k1 = (t + 1) << 5;
      async_b128(lA0[buf ^ 1], pa0 + k1);
      async_b128(lA1[buf ^ 1], pa1 + k1);
      async_b128(lB[buf ^ 1],  pb  + k1);
    }

    union { uint4 u[2]; v16bf v; } af, bf0, bf1, bf2, bf3;
    af.u[0]  = *(const uint4*)&As[buf][am][alo];
    af.u[1]  = *(const uint4*)&As[buf][am][alo + 8];
    bf0.u[0] = *(const uint4*)&Bs[buf][bn][bklo];
    bf0.u[1] = *(const uint4*)&Bs[buf][bn][bklo + 8];
    bf1.u[0] = *(const uint4*)&Bs[buf][16 + bn][bklo];
    bf1.u[1] = *(const uint4*)&Bs[buf][16 + bn][bklo + 8];
    bf2.u[0] = *(const uint4*)&Bs[buf][32 + bn][bklo];
    bf2.u[1] = *(const uint4*)&Bs[buf][32 + bn][bklo + 8];
    bf3.u[0] = *(const uint4*)&Bs[buf][48 + bn][bklo];
    bf3.u[1] = *(const uint4*)&Bs[buf][48 + bn][bklo + 8];

    acc0 = __builtin_amdgcn_wmma_f32_16x16x32_bf16(false, af.v, false, bf0.v,
                                                   (short)0, acc0, false, false);
    acc1 = __builtin_amdgcn_wmma_f32_16x16x32_bf16(false, af.v, false, bf1.v,
                                                   (short)0, acc1, false, false);
    acc2 = __builtin_amdgcn_wmma_f32_16x16x32_bf16(false, af.v, false, bf2.v,
                                                   (short)0, acc2, false, false);
    acc3 = __builtin_amdgcn_wmma_f32_16x16x32_bf16(false, af.v, false, bf3.v,
                                                   (short)0, acc3, false, false);

    wait_async0();        // our tile(t+1) DMA complete
    __syncthreads();      // everyone's DMA complete + LDS reads of buf done
  }

  // Epilogue: D layout — VGPR r: lanes0-15 -> M=r, lanes16-31 -> M=8+r, N=lane&15
  const int lm = lane & 15;
  const int hi = lane >> 4;
  #pragma unroll
  for (int r = 0; r < 8; ++r) {
    const int m = m0 + wm + r + hi * 8;
    if (m >= M) continue;
    v8f* accs[4] = { &acc0, &acc1, &acc2, &acc3 };
    #pragma unroll
    for (int j = 0; j < 4; ++j) {
      const int n = n0 + j * 16 + lm;
      if (n < N) {
        float v = (*accs[j])[r] + (bias ? bias[n] : 0.f);
        C[(size_t)m * ldc + n] = v;
        if (Csilu) Csilu[(size_t)m * ldc + n] = silu_f(v);
      }
    }
  }
}

// ---------------------------------------------------------------------------
// Weight f32 -> bf16 with K padding (zeros)
// ---------------------------------------------------------------------------
__global__ void cvt_w_kernel(const float* __restrict__ W, bf16_t* __restrict__ out,
                             int N, int Ks, int Kp)
{
  int t = blockIdx.x * blockDim.x + threadIdx.x;
  if (t >= N * Kp) return;
  int n = t / Kp, k = t - n * Kp;
  out[t] = (k < Ks) ? f2bf(W[(size_t)n * Ks + k]) : f2bf(0.f);
}

// ---------------------------------------------------------------------------
// Pack scalar features into bf16 X: cols [0,lenA) from sA, [lenA,lenA+lenB) from sB,
// skip [fin, fin+h) (written by vh kernel as vector norms), zero-pad the rest.
// ---------------------------------------------------------------------------
__global__ void pack_s_kernel(const float* __restrict__ sA, int lenA,
                              const float* __restrict__ sB, int lenB,
                              bf16_t* __restrict__ X, int Kp, int h, int M)
{
  int t = blockIdx.x * blockDim.x + threadIdx.x;
  if (t >= M * Kp) return;
  int m = t / Kp, k = t - m * Kp;
  int fin = lenA + lenB;
  if (k >= fin && k < fin + h) return;
  float v = 0.f;
  if (k < lenA)      v = sA[(size_t)m * lenA + k];
  else if (k < fin)  v = sB[(size_t)m * lenB + (k - lenA)];
  X[t] = f2bf(v);
}

// ---------------------------------------------------------------------------
// Edge gathers (message-stack GVP0 inputs)
// ---------------------------------------------------------------------------
__global__ void edge_gather_v_kernel(const float* __restrict__ vsrc, const float* __restrict__ xsrc,
                                     const float* __restrict__ vdst, const float* __restrict__ xdst,
                                     const int* __restrict__ src, const int* __restrict__ dst,
                                     int e0, int ec,
                                     float* __restrict__ evin /*[ec,33,3]*/,
                                     float* __restrict__ ed   /*[ec]*/)
{
  int t = blockIdx.x * blockDim.x + threadIdx.x;
  if (t >= ec * 33) return;
  int e = t / 33, slot = t - e * 33;
  int ge = e0 + e;
  int s = src[ge], d = dst[ge];
  float* o = evin + ((size_t)e * 33 + slot) * 3;
  if (slot < 16) {
    const float* p = vsrc + ((size_t)s * 16 + slot) * 3;
    o[0] = p[0]; o[1] = p[1]; o[2] = p[2];
  } else if (slot < 32) {
    const float* p = vdst + ((size_t)d * 16 + (slot - 16)) * 3;
    o[0] = p[0]; o[1] = p[1]; o[2] = p[2];
  } else {
    float x0 = xdst[(size_t)d * 3 + 0] - xsrc[(size_t)s * 3 + 0];
    float x1 = xdst[(size_t)d * 3 + 1] - xsrc[(size_t)s * 3 + 1];
    float x2 = xdst[(size_t)d * 3 + 2] - xsrc[(size_t)s * 3 + 2];
    float dn = sqrtf(x0 * x0 + x1 * x1 + x2 * x2);
    ed[e] = dn;
    float inv = 1.f / (dn + 1e-8f);
    o[0] = x0 * inv; o[1] = x1 * inv; o[2] = x2 * inv;
  }
}

__global__ void edge_gather_s_kernel(const float* __restrict__ ssrc, const float* __restrict__ sdst,
                                     const int* __restrict__ src, const int* __restrict__ dst,
                                     int e0, int ec, const float* __restrict__ ed,
                                     bf16_t* __restrict__ X, int Kp, int fin, int h)
{
  int t = blockIdx.x * blockDim.x + threadIdx.x;
  if (t >= ec * Kp) return;
  int e = t / Kp, k = t - e * Kp;
  int ge = e0 + e;
  float v;
  if (k < SDIM)            v = ssrc[(size_t)src[ge] * SDIM + k];
  else if (k < 2 * SDIM)   v = sdst[(size_t)dst[ge] * SDIM + (k - SDIM)];
  else if (k == 2 * SDIM)  v = ed[e];
  else if (k < fin + h)    return;   // vn region, written by vh kernel
  else                     v = 0.f;
  X[(size_t)e * Kp + k] = f2bf(v);
}

// ---------------------------------------------------------------------------
// vh = Wh @ v (per coordinate), vn = ||vh|| -> written as bf16 into X[:, fin:fin+h]
// vin may be the concat of two sources (vA [M,va,3], vB [M,vb,3]).
// ---------------------------------------------------------------------------
__global__ void gvp_vh_kernel(const float* __restrict__ vA, int va,
                              const float* __restrict__ vB, int vb,
                              const float* __restrict__ Wh,   // [h, va+vb]
                              float* __restrict__ vh,         // [M, h, 3]
                              bf16_t* __restrict__ X, int Kp, int fin,
                              int M, int h)
{
  extern __shared__ float sW[];
  const int vin = va + vb;
  for (int i = threadIdx.x; i < h * vin; i += blockDim.x) sW[i] = Wh[i];
  __syncthreads();
  int t = blockIdx.x * blockDim.x + threadIdx.x;
  if (t >= M * h) return;
  int m = t / h, hh = t - m * h;
  const float* w = sW + (size_t)hh * vin;
  float a0 = 0.f, a1 = 0.f, a2 = 0.f;
  const float* pA = vA + (size_t)m * va * 3;
  for (int v = 0; v < va; ++v) {
    float c = w[v];
    a0 += c * pA[v * 3 + 0]; a1 += c * pA[v * 3 + 1]; a2 += c * pA[v * 3 + 2];
  }
  if (vB) {
    const float* pB = vB + (size_t)m * vb * 3;
    for (int v = 0; v < vb; ++v) {
      float c = w[va + v];
      a0 += c * pB[v * 3 + 0]; a1 += c * pB[v * 3 + 1]; a2 += c * pB[v * 3 + 2];
    }
  }
  size_t o = ((size_t)m * h + hh) * 3;
  vh[o + 0] = a0; vh[o + 1] = a1; vh[o + 2] = a2;
  float vn = sqrtf(a0 * a0 + a1 * a1 + a2 * a2 + 1e-8f);
  X[(size_t)m * Kp + fin + hh] = f2bf(vn);
}

// ---------------------------------------------------------------------------
// vout[m,u,:] = (Wu[u,:] . vh[m,:,:]) * gate,  gate = (sigmoid?) (Wg[u,:].slin[m,:]+bg[u])
// ---------------------------------------------------------------------------
__global__ void gvp_out_kernel(const float* __restrict__ slin, int lds_, int fout,
                               const float* __restrict__ vh, int h,
                               const float* __restrict__ Wu, const float* __restrict__ Wg,
                               const float* __restrict__ bg,
                               float* __restrict__ vout, int M, int vo, int sigmoid_gate)
{
  extern __shared__ float sm[];
  float* sWu = sm;
  float* sWg = sm + vo * h;
  float* sbg = sWg + vo * fout;
  for (int i = threadIdx.x; i < vo * h;    i += blockDim.x) sWu[i] = Wu[i];
  for (int i = threadIdx.x; i < vo * fout; i += blockDim.x) sWg[i] = Wg[i];
  for (int i = threadIdx.x; i < vo;        i += blockDim.x) sbg[i] = bg[i];
  __syncthreads();
  int t = blockIdx.x * blockDim.x + threadIdx.x;
  if (t >= M * vo) return;
  int m = t / vo, u = t - m * vo;
  const float* sl = slin + (size_t)m * lds_;
  const float* wg = sWg + (size_t)u * fout;
  float g = sbg[u];
  for (int f = 0; f < fout; ++f) g += wg[f] * sl[f];
  if (sigmoid_gate) g = 1.f / (1.f + __expf(-g));
  const float* wu  = sWu + (size_t)u * h;
  const float* vhm = vh + (size_t)m * h * 3;
  float a0 = 0.f, a1 = 0.f, a2 = 0.f;
  for (int k = 0; k < h; ++k) {
    float c = wu[k];
    a0 += c * vhm[k * 3 + 0]; a1 += c * vhm[k * 3 + 1]; a2 += c * vhm[k * 3 + 2];
  }
  float* o = vout + ((size_t)m * vo + u) * 3;
  o[0] = a0 * g; o[1] = a1 * g; o[2] = a2 * g;
}

// ---------------------------------------------------------------------------
// Scatter-add of messages into destination node aggregates (scaled)
// ---------------------------------------------------------------------------
__global__ void scatter_kernel(const float* __restrict__ msg, const int* __restrict__ dst,
                               int e0, int ec, float* __restrict__ agg, int F, float scale)
{
  int t = blockIdx.x * blockDim.x + threadIdx.x;
  if (t >= ec * F) return;
  int e = t / F, f = t - e * F;
  int d = dst[e0 + e];
  __hip_atomic_fetch_add(&agg[(size_t)d * F + f], msg[(size_t)e * F + f] * scale,
                         __ATOMIC_RELAXED, __HIP_MEMORY_SCOPE_AGENT);
}

__global__ void add_inplace_kernel(float* __restrict__ d, const float* __restrict__ s, int n)
{
  int t = blockIdx.x * blockDim.x + threadIdx.x;
  if (t < n) d[t] += s[t];
}

// ---------------------------------------------------------------------------
// Host-side orchestration
// ---------------------------------------------------------------------------
struct GvpParam {
  const float *Wh, *Wu, *Ws, *bs, *Wg, *bg;
  bf16_t* Ws_bf;
  int v_in, v_out, f_in, f_out, h, Kpad;
};

static GvpParam take_gvp(void* const* d_in, int& idx, int v_in, int v_out, int f_in, int f_out)
{
  GvpParam g;
  g.Wh = (const float*)d_in[idx + 0];
  g.Wu = (const float*)d_in[idx + 1];
  g.Ws = (const float*)d_in[idx + 2];
  g.bs = (const float*)d_in[idx + 3];
  g.Wg = (const float*)d_in[idx + 4];
  g.bg = (const float*)d_in[idx + 5];
  idx += 6;
  g.v_in = v_in; g.v_out = v_out; g.f_in = f_in; g.f_out = f_out;
  g.h = v_in > v_out ? v_in : v_out;
  g.Kpad = ((f_in + g.h) + 31) / 32 * 32;
  g.Ws_bf = nullptr;
  return g;
}

struct Bump {
  char* base; size_t used, cap;
  void* take(size_t bytes) {
    size_t o = (used + 255) & ~(size_t)255;
    used = o + bytes;
    return base + o;
  }
};

static void apply_gvp(hipStream_t st, const GvpParam& g,
                      const float* sA, int lA, const float* sB, int lB,
                      const float* vA, int va, const float* vB, int vb,
                      bf16_t* X, float* vhbuf, float* slin, int lds_,
                      float* sout, float* vout, int M, int sigmoid)
{
  int tp = M * g.Kpad;
  pack_s_kernel<<<ceil_div(tp, 256), 256, 0, st>>>(sA, lA, sB, lB, X, g.Kpad, g.h, M);
  tp = M * g.h;
  gvp_vh_kernel<<<ceil_div(tp, 256), 256, (size_t)g.h * (va + vb) * 4, st>>>(
      vA, va, vB, vb, g.Wh, vhbuf, X, g.Kpad, lA + lB, M, g.h);
  dim3 gg(ceil_div(M, BM), ceil_div(g.f_out, BN));
  gemm_xwt_kernel<<<gg, 256, 0, st>>>(X, g.Ws_bf, g.bs, slin, sout, M, g.f_out, g.Kpad, lds_);
  tp = M * g.v_out;
  size_t sh = ((size_t)g.v_out * g.h + (size_t)g.v_out * g.f_out + g.v_out) * 4;
  gvp_out_kernel<<<ceil_div(tp, 256), 256, sh, st>>>(
      slin, lds_, g.f_out, vhbuf, g.h, g.Wu, g.Wg, g.bg, vout, M, g.v_out, sigmoid);
}

struct EdgeBufs {
  float *vin, *d, *vh, *slin, *sA, *sB, *vA, *vB;
  bf16_t* X;
};

static void run_msg(hipStream_t st, const GvpParam* stk,
                    const float* ssrc, const float* vsrc, const float* xsrc,
                    const float* sdst, const float* vdst, const float* xdst,
                    const int* src, const int* dst, int E,
                    const EdgeBufs& eb, float* agg_s, float* agg_v, int CH)
{
  const GvpParam& g0 = stk[0];
  for (int e0 = 0; e0 < E; e0 += CH) {
    int ec = E - e0 < CH ? E - e0 : CH;
    int tp = ec * 33;
    edge_gather_v_kernel<<<ceil_div(tp, 256), 256, 0, st>>>(
        vsrc, xsrc, vdst, xdst, src, dst, e0, ec, eb.vin, eb.d);
    tp = ec * g0.Kpad;
    edge_gather_s_kernel<<<ceil_div(tp, 256), 256, 0, st>>>(
        ssrc, sdst, src, dst, e0, ec, eb.d, eb.X, g0.Kpad, g0.f_in, g0.h);
    // GVP 0 (inputs already packed / gathered)
    tp = ec * g0.h;
    gvp_vh_kernel<<<ceil_div(tp, 256), 256, (size_t)g0.h * g0.v_in * 4, st>>>(
        eb.vin, g0.v_in, nullptr, 0, g0.Wh, eb.vh, eb.X, g0.Kpad, g0.f_in, ec, g0.h);
    dim3 gg(ceil_div(ec, BM), ceil_div(g0.f_out, BN));
    gemm_xwt_kernel<<<gg, 256, 0, st>>>(eb.X, g0.Ws_bf, g0.bs, eb.slin, eb.sA,
                                        ec, g0.f_out, g0.Kpad, SDIM);
    tp = ec * g0.v_out;
    size_t sh = ((size_t)g0.v_out * g0.h + (size_t)g0.v_out * g0.f_out + g0.v_out) * 4;
    gvp_out_kernel<<<ceil_div(tp, 256), 256, sh, st>>>(
        eb.slin, SDIM, g0.f_out, eb.vh, g0.h, g0.Wu, g0.Wg, g0.bg, eb.vA, ec, g0.v_out, 1);
    // GVP 1
    apply_gvp(st, stk[1], eb.sA, SDIM, nullptr, 0, eb.vA, VDIM, nullptr, 0,
              eb.X, eb.vh, eb.slin, SDIM, eb.sB, eb.vB, ec, 1);
    // GVP 2
    apply_gvp(st, stk[2], eb.sB, SDIM, nullptr, 0, eb.vB, VDIM, nullptr, 0,
              eb.X, eb.vh, eb.slin, SDIM, eb.sA, eb.vA, ec, 1);
    // Scatter-add (with 1/MSG_NORM folded in)
    scatter_kernel<<<ceil_div(ec * SDIM, 256), 256, 0, st>>>(eb.sA, dst, e0, ec, agg_s, SDIM, 0.1f);
    scatter_kernel<<<ceil_div(ec * 48, 256), 256, 0, st>>>(eb.vA, dst, e0, ec, agg_v, 48, 0.1f);
  }
}

extern "C" void kernel_launch(void* const* d_in, const int* in_sizes, int n_in,
                              void* d_out, int out_size, void* d_ws, size_t ws_size,
                              hipStream_t stream)
{
  (void)in_sizes; (void)n_in; (void)out_size;
  const float* in_lig_s = (const float*)d_in[0];
  const float* in_lig_x = (const float*)d_in[1];
  const float* in_lig_v = (const float*)d_in[2];
  const float* in_kp_s  = (const float*)d_in[3];
  const float* in_kp_x  = (const float*)d_in[4];
  const float* in_kp_v  = (const float*)d_in[5];
  const int* ll_src = (const int*)d_in[6];
  const int* ll_dst = (const int*)d_in[7];
  const int* kl_src = (const int*)d_in[8];
  const int* kl_dst = (const int*)d_in[9];

  // ---- parameter walk (dict insertion order) ----
  int idx = 10;
  GvpParam convs[4][8];   // [0..2]=msg_ll, [3..5]=msg_kl, [6..7]=upd
  for (int c = 0; c < 4; ++c) {
    for (int i = 0; i < 3; ++i)
      convs[c][i] = take_gvp(d_in, idx, i == 0 ? 33 : 16, 16, i == 0 ? 513 : 256, 256);
    for (int i = 0; i < 3; ++i)
      convs[c][3 + i] = take_gvp(d_in, idx, i == 0 ? 33 : 16, 16, i == 0 ? 513 : 256, 256);
    for (int i = 0; i < 2; ++i)
      convs[c][6 + i] = take_gvp(d_in, idx, i == 0 ? 32 : 16, 16, i == 0 ? 512 : 256, 256);
  }
  GvpParam noise[3];
  noise[0] = take_gvp(d_in, idx, 16, 16, 256, 256);
  noise[1] = take_gvp(d_in, idx, 16, 16, 256, 256);
  noise[2] = take_gvp(d_in, idx, 16, 1, 256, 64);
  const float* WlinW = (const float*)d_in[idx++];  // [128,64]
  const float* Wlinb = (const float*)d_in[idx++];  // [128]

  // ---- workspace layout ----
  Bump ws{(char*)d_ws, 0, ws_size};
  float* lig_s = (float*)ws.take((size_t)NLIG * SDIM * 4);
  float* lig_v = (float*)ws.take((size_t)NLIG * 48 * 4);
  float* agg_s = (float*)ws.take((size_t)NLIG * SDIM * 4);
  float* agg_v = (float*)ws.take((size_t)NLIG * 48 * 4);
  bf16_t* nX   = (bf16_t*)ws.take((size_t)NLIG * 576 * 2);
  float* nvh   = (float*)ws.take((size_t)NLIG * 32 * 3 * 4);
  float* nslin = (float*)ws.take((size_t)NLIG * SDIM * 4);
  float* nsA   = (float*)ws.take((size_t)NLIG * SDIM * 4);
  float* nsB   = (float*)ws.take((size_t)NLIG * SDIM * 4);
  float* nvA   = (float*)ws.take((size_t)NLIG * 48 * 4);
  float* nvB   = (float*)ws.take((size_t)NLIG * 48 * 4);
  bf16_t* Xlin = (bf16_t*)ws.take((size_t)NLIG * 64 * 2);
  // bf16 weight buffers
  for (int c = 0; c < 4; ++c)
    for (int i = 0; i < 8; ++i)
      convs[c][i].Ws_bf = (bf16_t*)ws.take((size_t)convs[c][i].f_out * convs[c][i].Kpad * 2);
  for (int i = 0; i < 3; ++i)
    noise[i].Ws_bf = (bf16_t*)ws.take((size_t)noise[i].f_out * noise[i].Kpad * 2);
  bf16_t* Wlin_bf = (bf16_t*)ws.take((size_t)128 * 64 * 2);

  // ---- adaptive edge chunk from remaining workspace ----
  const long long perEdge = 396 + 4 + 1152 + 396 + 1024 + 1024 + 1024 + 192 + 192; // 5404 B
  long long rem = (long long)ws_size - (long long)ws.used - 16 * 256;
  int CH = 16384;
  if (rem > 0) {
    long long c = rem / perEdge;
    if (c < CH) CH = (int)c;
  }
  CH = (CH / 64) * 64;
  if (CH < 64) CH = 64;

  EdgeBufs eb;
  eb.vin  = (float*)ws.take((size_t)CH * 33 * 3 * 4);
  eb.d    = (float*)ws.take((size_t)CH * 4);
  eb.X    = (bf16_t*)ws.take((size_t)CH * 576 * 2);
  eb.vh   = (float*)ws.take((size_t)CH * 33 * 3 * 4);
  eb.slin = (float*)ws.take((size_t)CH * SDIM * 4);
  eb.sA   = (float*)ws.take((size_t)CH * SDIM * 4);
  eb.sB   = (float*)ws.take((size_t)CH * SDIM * 4);
  eb.vA   = (float*)ws.take((size_t)CH * 48 * 4);
  eb.vB   = (float*)ws.take((size_t)CH * 48 * 4);

  // ---- convert weights to padded bf16 ----
  for (int c = 0; c < 4; ++c)
    for (int i = 0; i < 8; ++i) {
      GvpParam& g = convs[c][i];
      int tp = g.f_out * g.Kpad;
      cvt_w_kernel<<<ceil_div(tp, 256), 256, 0, stream>>>(g.Ws, g.Ws_bf, g.f_out,
                                                          g.f_in + g.h, g.Kpad);
    }
  for (int i = 0; i < 3; ++i) {
    GvpParam& g = noise[i];
    int tp = g.f_out * g.Kpad;
    cvt_w_kernel<<<ceil_div(tp, 256), 256, 0, stream>>>(g.Ws, g.Ws_bf, g.f_out,
                                                        g.f_in + g.h, g.Kpad);
  }
  cvt_w_kernel<<<ceil_div(128 * 64, 256), 256, 0, stream>>>(WlinW, Wlin_bf, 128, 64, 64);

  // ---- init node state ----
  hipMemcpyAsync(lig_s, in_lig_s, (size_t)NLIG * SDIM * 4, hipMemcpyDeviceToDevice, stream);
  hipMemcpyAsync(lig_v, in_lig_v, (size_t)NLIG * 48 * 4, hipMemcpyDeviceToDevice, stream);

  // ---- conv layers ----
  for (int c = 0; c < 4; ++c) {
    hipMemsetAsync(agg_s, 0, (size_t)NLIG * SDIM * 4, stream);
    hipMemsetAsync(agg_v, 0, (size_t)NLIG * 48 * 4, stream);
    run_msg(stream, &convs[c][0], lig_s, lig_v, in_lig_x, lig_s, lig_v, in_lig_x,
            ll_src, ll_dst, ELL, eb, agg_s, agg_v, CH);
    run_msg(stream, &convs[c][3], in_kp_s, in_kp_v, in_kp_x, lig_s, lig_v, in_lig_x,
            kl_src, kl_dst, EKL, eb, agg_s, agg_v, CH);
    // update stack
    apply_gvp(stream, convs[c][6], lig_s, SDIM, agg_s, SDIM, lig_v, VDIM, agg_v, VDIM,
              nX, nvh, nslin, SDIM, nsA, nvA, NLIG, 1);
    apply_gvp(stream, convs[c][7], nsA, SDIM, nullptr, 0, nvA, VDIM, nullptr, 0,
              nX, nvh, nslin, SDIM, nsB, nvB, NLIG, 1);
    add_inplace_kernel<<<ceil_div(NLIG * SDIM, 256), 256, 0, stream>>>(lig_s, nsB, NLIG * SDIM);
    add_inplace_kernel<<<ceil_div(NLIG * 48, 256), 256, 0, stream>>>(lig_v, nvB, NLIG * 48);
  }

  // ---- noise head ----
  apply_gvp(stream, noise[0], lig_s, SDIM, nullptr, 0, lig_v, VDIM, nullptr, 0,
            nX, nvh, nslin, SDIM, nsA, nvA, NLIG, 1);
  apply_gvp(stream, noise[1], nsA, SDIM, nullptr, 0, nvA, VDIM, nullptr, 0,
            nX, nvh, nslin, SDIM, nsB, nvB, NLIG, 1);
  // last GVP: f_out=64 (ldc=64), identity vector gate
  apply_gvp(stream, noise[2], nsB, SDIM, nullptr, 0, nvB, VDIM, nullptr, 0,
            nX, nvh, nslin, 64, nsA, nvA, NLIG, 0);

  // final linear: [NLIG,64](silu) @ Wlin^T + b -> d_out[0 : NLIG*128)
  pack_s_kernel<<<ceil_div(NLIG * 64, 256), 256, 0, stream>>>(nsA, 64, nullptr, 0,
                                                              Xlin, 64, 0, NLIG);
  {
    dim3 gg(ceil_div(NLIG, BM), ceil_div(128, BN));
    gemm_xwt_kernel<<<gg, 256, 0, stream>>>(Xlin, Wlin_bf, Wlinb, (float*)d_out, nullptr,
                                            NLIG, 128, 64, 128);
  }
  // vector output [NLIG,1,3] appended after scalar output
  hipMemcpyAsync((float*)d_out + (size_t)NLIG * 128, nvA, (size_t)NLIG * 3 * 4,
                 hipMemcpyDeviceToDevice, stream);
}
